// StandardAttention_27642409517612
// MI455X (gfx1250) — matrix operand order repaired and checked by reading, not compiled, not run on hardware
//
#include <hip/hip_runtime.h>
#include <cstdint>

typedef _Float16 f16;
typedef __attribute__((ext_vector_type(16))) _Float16 v16h;
typedef __attribute__((ext_vector_type(8)))  _Float16 v8h;
typedef __attribute__((ext_vector_type(4)))  _Float16 v4h;
typedef __attribute__((ext_vector_type(8)))  float    v8f;

#define B_   2
#define S_   2048
#define D_   1024
#define H_   16
#define DH_  64
#define MTOT (B_ * S_)     // 4096
#define NQKV (3 * D_)      // 3072

// ---------------------------------------------------------------------------
// WMMA wrapper: D(f32 16x16) = A(f16 16x32) * B(f16 32x16) + C
// ---------------------------------------------------------------------------
__device__ __forceinline__ v8f wmma16(v16h a, v16h b, v8f c) {
    return __builtin_amdgcn_wmma_f32_16x16x32_f16(
        /*neg_a=*/false, a, /*neg_b=*/false, b,
        /*c_mod=*/(short)0, c, /*reuse_a=*/false, /*reuse_b=*/false);
}

// A-fragment loader (16x32 f16). Lane owns one row; lanes 0-15
// hold K {0..15}, lanes 16-31 hold K {16..31}. `hf` = lane>=16.
__device__ __forceinline__ v16h ldA(const f16* row, int hf) {
    v8h lo = *(const v8h*)(row + 16 * hf);
    v8h hi = *(const v8h*)(row + 16 * hf + 8);
    v16h a;
#pragma unroll
    for (int i = 0; i < 8; ++i) { a[i] = lo[i]; a[i + 8] = hi[i]; }
    return a;
}

// ---------------------------------------------------------------------------
// Async global->LDS staging via CDNA5 GLOBAL_LOAD_ASYNC_TO_LDS_B128.
// VDST = wave-relative LDS byte address (low 32 bits of the generic pointer),
// VADDR = 64-bit global address (GV mode). Tracked by ASYNCcnt.
// ---------------------------------------------------------------------------
__device__ __forceinline__ void stage16_async(const f16* gsrc, f16* ldst) {
    uint32_t lds_off = (uint32_t)(uintptr_t)ldst;
    asm volatile("global_load_async_to_lds_b128 %0, %1, off"
                 :
                 : "v"(lds_off), "v"(gsrc)
                 : "memory");
}

__device__ __forceinline__ void wait_async_lds() {
    asm volatile("s_wait_asynccnt 0" ::: "memory");
}

// ---------------------------------------------------------------------------
// f32 -> f16 conversion (vectorized)
// ---------------------------------------------------------------------------
__global__ void __launch_bounds__(256)
cvt_kernel(const float4* __restrict__ src, v4h* __restrict__ dst) {
    int i = blockIdx.x * 256 + threadIdx.x;
    float4 v = src[i];
    v4h h;
    h[0] = (f16)v.x; h[1] = (f16)v.y; h[2] = (f16)v.z; h[3] = (f16)v.w;
    dst[i] = h;
}

// ---------------------------------------------------------------------------
// QKV projection: qkv[m][n] = sum_d x[m][d] * qkv_w[n][d] + qkv_b[n]
// Q,K written [B][H][S][DH]; V written TRANSPOSED [B][H][DH][S] so that the
// attention P@V B-fragments become contiguous loads.
// Block = 8 waves, tile 128(M) x 64(N); weight tile staged in LDS (async,
// double-buffered) and shared by all 8 waves.
// ---------------------------------------------------------------------------
__global__ void __launch_bounds__(256)
qkv_gemm_kernel(const f16* __restrict__ xh, const f16* __restrict__ wh,
                const float* __restrict__ bias,
                f16* __restrict__ Qh, f16* __restrict__ Kh, f16* __restrict__ Vt) {
    __shared__ __align__(128) f16 bB[2][64][32];   // 2 x 4KB weight tiles

    const int tid  = threadIdx.x;
    const int lane = tid & 31;
    const int wv   = tid >> 5;
    const int hf   = lane >> 4;
    const int r    = lane & 15;
    const int m0   = blockIdx.x * 128 + wv * 16;
    const int n0   = blockIdx.y * 64;

    // cooperative B-tile staging: thread -> (row 0..63, 8-col group 0..3)
    const int brow = tid >> 2;
    const int bcol = (tid & 3) * 8;
    const f16* bsrc = wh + (size_t)(n0 + brow) * D_ + bcol;

    v8f acc0 = {}, acc1 = {}, acc2 = {}, acc3 = {};
    const f16* xrow = xh + (size_t)(m0 + r) * D_;
    const f16* lb   = &bB[0][r][16 * hf];          // lane's B-read base (t=0, buf 0)

    stage16_async(bsrc, &bB[0][brow][bcol]);       // k0 = 0
#pragma unroll 1
    for (int i = 0; i < D_ / 32; ++i) {
        const int k0  = i * 32;
        const int cur = i & 1;
        wait_async_lds();
        __syncthreads();                           // tile i ready, prev buf free
        if (i + 1 < D_ / 32)
            stage16_async(bsrc + (k0 + 32), &bB[1 - cur][brow][bcol]);

        v16h a = ldA(xrow + k0, hf);
        const f16* lbc = lb + cur * (64 * 32);     // select buffer
        acc0 = wmma16(a, *(const v16h*)(lbc +  0 * 32), acc0);
        acc1 = wmma16(a, *(const v16h*)(lbc + 16 * 32), acc1);
        acc2 = wmma16(a, *(const v16h*)(lbc + 32 * 32), acc2);
        acc3 = wmma16(a, *(const v16h*)(lbc + 48 * 32), acc3);
    }

#pragma unroll
    for (int t = 0; t < 4; ++t) {
        v8f acc = (t == 0) ? acc0 : (t == 1) ? acc1 : (t == 2) ? acc2 : acc3;
        int n = n0 + t * 16 + r;
        float bn = bias[n];
        int sel = n >> 10;            // 0=Q 1=K 2=V  (uniform per block)
        int di  = n & 1023;
        int h   = di >> 6;
        int dh  = di & 63;
        if (sel == 2) {
#pragma unroll
            for (int j = 0; j < 8; ++j) {
                int m  = m0 + j + 8 * hf;
                int bb = m >> 11;
                int ss = m & 2047;
                Vt[(((size_t)bb * H_ + h) * DH_ + dh) * S_ + ss] = (f16)(acc[j] + bn);
            }
        } else {
            f16* dst = (sel == 0) ? Qh : Kh;
#pragma unroll
            for (int j = 0; j < 8; ++j) {
                int m  = m0 + j + 8 * hf;
                int bb = m >> 11;
                int ss = m & 2047;
                dst[(((size_t)bb * H_ + h) * S_ + ss) * DH_ + dh] = (f16)(acc[j] + bn);
            }
        }
    }
}

// ---------------------------------------------------------------------------
// Flash attention: one wave per (b, h, 16-query tile). Key tiles of 32.
// V is pre-transposed [B][H][DH][S] -> every V B-fragment is one contiguous
// 32B load, same shape as the K-fragment loads.
// ---------------------------------------------------------------------------
__global__ void __launch_bounds__(256)
attn_kernel(const f16* __restrict__ Qh, const f16* __restrict__ Kh,
            const f16* __restrict__ Vt, f16* __restrict__ attnh) {
    __shared__ __align__(128) f16 pbuf[8][16][32];

    const int lane = threadIdx.x & 31;
    const int wv   = threadIdx.x >> 5;
    const int hf   = lane >> 4;
    const int r    = lane & 15;
    const int gw   = blockIdx.x * 8 + wv;      // 0 .. B*H*(S/16)-1
    const int qt   = gw & 127;                 // S/16 = 128
    const int bh   = gw >> 7;                  // 0..31 (= b*H + h)
    const int q0   = qt * 16;

    const f16* qbase = Qh + (size_t)bh * S_ * DH_;
    const f16* kbase = Kh + (size_t)bh * S_ * DH_;
    const f16* vbase = Vt + (size_t)bh * DH_ * S_;

    v16h qa0 = ldA(qbase + (size_t)(q0 + r) * DH_ + 0, hf);
    v16h qa1 = ldA(qbase + (size_t)(q0 + r) * DH_ + 32, hf);

    v8f O0 = {}, O1 = {}, O2 = {}, O3 = {};
    float mrun[8], lrun[8];
#pragma unroll
    for (int j = 0; j < 8; ++j) { mrun[j] = -1e30f; lrun[j] = 0.f; }

    const int ntiles = (q0 + 47) >> 5;         // ceil((q0+16)/32)
    const float sc_mul = 0.125f;               // 1/sqrt(DH)

#pragma unroll 1
    for (int t = 0; t < ntiles; ++t) {
        const int kb0 = t * 32;

        // ---- S = Q @ K^T (16 queries x 32 keys) -----------------------
        v8f s0 = {}, s1 = {};
        {
            const f16* k0p = kbase + (size_t)(kb0 + r) * DH_ + 16 * hf;
            v16h b00 = *(const v16h*)(k0p + 0);
            v16h b01 = *(const v16h*)(k0p + 32);
            s0 = wmma16(qa0, b00, s0);
            s0 = wmma16(qa1, b01, s0);
            const f16* k1p = kbase + (size_t)(kb0 + 16 + r) * DH_ + 16 * hf;
            v16h b10 = *(const v16h*)(k1p + 0);
            v16h b11 = *(const v16h*)(k1p + 32);
            s1 = wmma16(qa0, b10, s1);
            s1 = wmma16(qa1, b11, s1);
        }

        // ---- online softmax (row stats at j = row - 8*hf) -------------
        const int k_idx0 = kb0 + r;
        const int k_idx1 = kb0 + 16 + r;
        float scale[8];
#pragma unroll
        for (int j = 0; j < 8; ++j) {
            int q_idx = q0 + j + 8 * hf;
            float a = (k_idx0 <= q_idx) ? s0[j] * sc_mul : -1e30f;
            float b = (k_idx1 <= q_idx) ? s1[j] * sc_mul : -1e30f;
            float tm = fmaxf(a, b);
#pragma unroll
            for (int o = 1; o < 16; o <<= 1) tm = fmaxf(tm, __shfl_xor(tm, o, 32));
            float mn = fmaxf(mrun[j], tm);
            scale[j] = __expf(mrun[j] - mn);
            mrun[j] = mn;
            a = __expf(a - mn);
            b = __expf(b - mn);
            float ts = a + b;
#pragma unroll
            for (int o = 1; o < 16; o <<= 1) ts += __shfl_xor(ts, o, 32);
            lrun[j] = lrun[j] * scale[j] + ts;
            pbuf[wv][j + 8 * hf][r]      = (f16)a;   // P in C-layout -> LDS
            pbuf[wv][j + 8 * hf][16 + r] = (f16)b;
        }

#pragma unroll
        for (int j = 0; j < 8; ++j) {
            O0[j] *= scale[j]; O1[j] *= scale[j];
            O2[j] *= scale[j]; O3[j] *= scale[j];
        }

        asm volatile("s_wait_dscnt 0" ::: "memory");
        v16h pa = ldA(&pbuf[wv][r][0], hf);     // P as A-fragment (16x32)

        // ---- O += P @ V : 4 dh-blocks, each one contiguous 32B load ----
        const f16* vp = vbase + (size_t)r * S_ + kb0 + 16 * hf;
        O0 = wmma16(pa, *(const v16h*)(vp +  0 * S_), O0);
        O1 = wmma16(pa, *(const v16h*)(vp + 16 * S_), O1);
        O2 = wmma16(pa, *(const v16h*)(vp + 32 * S_), O2);
        O3 = wmma16(pa, *(const v16h*)(vp + 48 * S_), O3);
    }

    // ---- finalize: divide by row sums, write [b, s, h*DH + dh] --------
    const int b_i = bh >> 4, h_i = bh & 15;
#pragma unroll
    for (int j = 0; j < 8; ++j) {
        float inv = 1.0f / lrun[j];
        int m  = b_i * S_ + q0 + j + 8 * hf;
        int c0 = h_i * DH_ + r;
        attnh[(size_t)m * D_ + c0]      = (f16)(O0[j] * inv);
        attnh[(size_t)m * D_ + c0 + 16] = (f16)(O1[j] * inv);
        attnh[(size_t)m * D_ + c0 + 32] = (f16)(O2[j] * inv);
        attnh[(size_t)m * D_ + c0 + 48] = (f16)(O3[j] * inv);
    }
}

// ---------------------------------------------------------------------------
// Output projection: out[m][n] = sum_d attn[m][d] * out_w[n][d] + out_b[n]
// Same LDS-staged scheme as the QKV GEMM; fp32 result to d_out.
// ---------------------------------------------------------------------------
__global__ void __launch_bounds__(256)
out_gemm_kernel(const f16* __restrict__ ah, const f16* __restrict__ wh,
                const float* __restrict__ bias, float* __restrict__ out) {
    __shared__ __align__(128) f16 bB[2][64][32];

    const int tid  = threadIdx.x;
    const int lane = tid & 31;
    const int wv   = tid >> 5;
    const int hf   = lane >> 4;
    const int r    = lane & 15;
    const int m0   = blockIdx.x * 128 + wv * 16;
    const int n0   = blockIdx.y * 64;

    const int brow = tid >> 2;
    const int bcol = (tid & 3) * 8;
    const f16* bsrc = wh + (size_t)(n0 + brow) * D_ + bcol;

    v8f acc0 = {}, acc1 = {}, acc2 = {}, acc3 = {};
    const f16* arow = ah + (size_t)(m0 + r) * D_;
    const f16* lb   = &bB[0][r][16 * hf];

    stage16_async(bsrc, &bB[0][brow][bcol]);
#pragma unroll 1
    for (int i = 0; i < D_ / 32; ++i) {
        const int k0  = i * 32;
        const int cur = i & 1;
        wait_async_lds();
        __syncthreads();
        if (i + 1 < D_ / 32)
            stage16_async(bsrc + (k0 + 32), &bB[1 - cur][brow][bcol]);

        v16h a = ldA(arow + k0, hf);
        const f16* lbc = lb + cur * (64 * 32);
        acc0 = wmma16(a, *(const v16h*)(lbc +  0 * 32), acc0);
        acc1 = wmma16(a, *(const v16h*)(lbc + 16 * 32), acc1);
        acc2 = wmma16(a, *(const v16h*)(lbc + 32 * 32), acc2);
        acc3 = wmma16(a, *(const v16h*)(lbc + 48 * 32), acc3);
    }

#pragma unroll
    for (int t = 0; t < 4; ++t) {
        v8f acc = (t == 0) ? acc0 : (t == 1) ? acc1 : (t == 2) ? acc2 : acc3;
        int n = n0 + t * 16 + r;
        float bn = bias[n];
#pragma unroll
        for (int j = 0; j < 8; ++j) {
            int m = m0 + j + 8 * hf;
            out[(size_t)m * D_ + n] = acc[j] + bn;
        }
    }
}

// ---------------------------------------------------------------------------
// Launch
// ---------------------------------------------------------------------------
extern "C" void kernel_launch(void* const* d_in, const int* in_sizes, int n_in,
                              void* d_out, int out_size, void* d_ws, size_t ws_size,
                              hipStream_t stream) {
    const float* x     = (const float*)d_in[0];
    const float* qkv_w = (const float*)d_in[1];
    const float* qkv_b = (const float*)d_in[2];
    const float* out_w = (const float*)d_in[3];
    const float* out_b = (const float*)d_in[4];

    char* ws = (char*)d_ws;
    f16* xh  = (f16*)(ws + 0);               //  8 MB: x f16      [4096][1024]
    f16* wqh = (f16*)(ws + 8388608);         //  6 MB: qkv_w f16  [3072][1024]
    f16* woh = (f16*)(ws + 14680064);        //  2 MB: out_w f16  [1024][1024]
    f16* Qh  = (f16*)(ws + 16777216);        //  8 MB: Q f16 [B][H][S][DH]
    f16* Kh  = (f16*)(ws + 25165824);        //  8 MB: K f16 [B][H][S][DH]
    f16* Vt  = (f16*)(ws + 33554432);        //  8 MB: V f16 [B][H][DH][S] (transposed)
    f16* ah  = (f16*)(ws + 41943040);        //  8 MB: attn out f16 [4096][1024]

    cvt_kernel<<<(MTOT * D_ / 4) / 256, 256, 0, stream>>>((const float4*)x,     (v4h*)xh);
    cvt_kernel<<<(NQKV * D_ / 4) / 256, 256, 0, stream>>>((const float4*)qkv_w, (v4h*)wqh);
    cvt_kernel<<<(D_ * D_ / 4) / 256,   256, 0, stream>>>((const float4*)out_w, (v4h*)woh);

    qkv_gemm_kernel<<<dim3(MTOT / 128, NQKV / 64), 256, 0, stream>>>(
        xh, wqh, qkv_b, Qh, Kh, Vt);

    attn_kernel<<<(B_ * H_ * (S_ / 16)) / 8, 256, 0, stream>>>(Qh, Kh, Vt, ah);

    out_gemm_kernel<<<dim3(MTOT / 128, D_ / 64), 256, 0, stream>>>(
        ah, woh, out_b, (float*)d_out);
}
